// MixedAttentionModule_33844342293170
// MI455X (gfx1250) — matrix operand order repaired and checked by reading, not compiled, not run on hardware
//
#include <hip/hip_runtime.h>
#include <cstdint>
#include <cstddef>

// ---------------------------------------------------------------------------
// MixedAttentionModule for MI455X (gfx1250, wave32, WMMA).
//
// Shapes: B=32, n=1024 (32x32), C=768, NH=12, HD=64, keys/values n2=256, FF*C=3072.
// All GEMMs run on v_wmma_f32_16x16x32_f16 (f16 in, f32 accumulate).
// GEMM wave tile is 32x64: two A fragments share each B fragment
// (8 WMMA per 12 memory ops per k-step, 2x B reuse vs 16x64).
// Attention is flash-style over 256 keys (2 slabs x 128, chunks of 32),
// K / V^T staged in LDS, P transposed wave-locally through LDS.
// ---------------------------------------------------------------------------

#define Bsz 32
#define HtW 32
#define NTOK 1024
#define NTOK2 256
#define Cdim 768
#define NHEAD 12
#define HD 64
#define FFC 3072
#define EPSV 1e-5f

typedef _Float16 half_t;
typedef __attribute__((ext_vector_type(8)))  _Float16 v8h;
typedef __attribute__((ext_vector_type(16))) _Float16 v16h;
typedef __attribute__((ext_vector_type(8)))  float    v8f;

static __device__ __forceinline__ v8f vzero8f() {
  v8f z;
#pragma unroll
  for (int i = 0; i < 8; ++i) z[i] = 0.0f;
  return z;
}

// Assemble a 16-half fragment from two contiguous 8-half (16B) loads.
static __device__ __forceinline__ v16h make_frag(const half_t* p0, const half_t* p1) {
  v8h lo = *(const v8h*)p0;
  v8h hi = *(const v8h*)p1;
  return __builtin_shufflevector(lo, hi, 0, 1, 2, 3, 4, 5, 6, 7,
                                 8, 9, 10, 11, 12, 13, 14, 15);
}

// ---------------------------------------------------------------------------
// f32 -> f16 flat convert (weight staging; weights then live in L2).
// ---------------------------------------------------------------------------
__global__ void f32_to_f16_kernel(const float* __restrict__ a,
                                  half_t* __restrict__ o, int n) {
  int i = blockIdx.x * 256 + threadIdx.x;
  if (i < n) o[i] = (half_t)a[i];
}

// ---------------------------------------------------------------------------
// LayerNorm over C=768, one block (256 thr, 8 waves) per (b, token) row.
// Output f16 for WMMA consumption.
// ---------------------------------------------------------------------------
__global__ __launch_bounds__(256)
void ln_kernel(const float* __restrict__ in, const float* __restrict__ g,
               const float* __restrict__ bb, half_t* __restrict__ out) {
  const int row = blockIdx.x;
  const float* p = in + (size_t)row * Cdim;
  const int t = threadIdx.x;
  float x0 = p[t], x1 = p[t + 256], x2 = p[t + 512];
  float s = x0 + x1 + x2;
  float q = x0 * x0 + x1 * x1 + x2 * x2;
#pragma unroll
  for (int m = 1; m < 32; m <<= 1) {
    s += __shfl_xor(s, m, 32);
    q += __shfl_xor(q, m, 32);
  }
  __shared__ float ss[8], sq[8];
  const int wave = t >> 5, lane = t & 31;
  if (lane == 0) { ss[wave] = s; sq[wave] = q; }
  __syncthreads();
  if (t == 0) {
    float S = 0.f, Q = 0.f;
#pragma unroll
    for (int i = 0; i < 8; ++i) { S += ss[i]; Q += sq[i]; }
    ss[0] = S; sq[0] = Q;
  }
  __syncthreads();
  const float mean = ss[0] * (1.0f / Cdim);
  const float var  = sq[0] * (1.0f / Cdim) - mean * mean;
  const float inv  = rsqrtf(var + EPSV);
  half_t* o = out + (size_t)row * Cdim;
  o[t]       = (half_t)((x0 - mean) * inv * g[t]       + bb[t]);
  o[t + 256] = (half_t)((x1 - mean) * inv * g[t + 256] + bb[t + 256]);
  o[t + 512] = (half_t)((x2 - mean) * inv * g[t + 512] + bb[t + 512]);
}

// ---------------------------------------------------------------------------
// Depthwise 3x3 conv (pad 1, stride S) + bias + ELU + BatchNorm(eval) -> f16
// xn: (B, 1024, C) f16, spatial token = iy*32+ix.  out: (B, n_out, C) f16.
// Threads: c fastest -> fully coalesced.
// ---------------------------------------------------------------------------
template <int S>
__global__ __launch_bounds__(256)
void dw_kernel(const half_t* __restrict__ xn, const float* __restrict__ w,
               const float* __restrict__ bias, const float* __restrict__ g,
               const float* __restrict__ bb, const float* __restrict__ mm,
               const float* __restrict__ vv, half_t* __restrict__ out) {
  const int c  = blockIdx.x * 256 + threadIdx.x;      // < 768
  const int op = blockIdx.y;
  const int b  = blockIdx.z;
  const int Wo = (S == 1) ? 32 : 16;
  const int oy = op / Wo, ox = op % Wo;
  float acc = 0.0f;
#pragma unroll
  for (int ky = 0; ky < 3; ++ky) {
    const int iy = oy * S - 1 + ky;
    if ((unsigned)iy < 32u) {
#pragma unroll
      for (int kx = 0; kx < 3; ++kx) {
        const int ix = ox * S - 1 + kx;
        if ((unsigned)ix < 32u) {
          acc += (float)xn[((size_t)b * NTOK + iy * 32 + ix) * Cdim + c] *
                 w[c * 9 + ky * 3 + kx];
        }
      }
    }
  }
  acc += bias[c];
  acc = acc > 0.0f ? acc : (expf(acc) - 1.0f);                       // ELU
  acc = (acc - mm[c]) * rsqrtf(vv[c] + EPSV) * g[c] + bb[c];         // BN
  out[((size_t)b * (Wo * Wo) + op) * Cdim + c] = (half_t)acc;
}

// ---------------------------------------------------------------------------
// Generic WMMA GEMM: D(MxN) = A(MxK) @ W^T + bias, W stored (N,K) row-major f16.
// Block 256 thr = 8 waves; block tile 256x64; wave tile 32x64:
//   2 A-fragments x 4 B-fragments -> 8 WMMA per 32-wide k-step.
// A-fragment: lane m=L%16, two 16B loads per 32-wide k-step (ISA 16b A layout).
// B-fragment: lane n=L%16, two 16B loads per k-step (ISA 16b B layout).
// Epilogues: 0 = bias->f16, 1 = GELU(bias)->f16, 2 = bias+res->f32.
// ---------------------------------------------------------------------------
template <int EPI>
__global__ __launch_bounds__(256)
void gemm_wmma_kernel(const half_t* __restrict__ A, const half_t* __restrict__ W,
                      const float* __restrict__ bias, const float* __restrict__ res,
                      half_t* __restrict__ outH, float* __restrict__ outF,
                      int M, int N, int K) {
  const int lane  = threadIdx.x & 31;
  const int wave  = threadIdx.x >> 5;
  const int mLoc  = lane & 15;
  const int hi    = lane >> 4;
  const int mBase = blockIdx.y * 256 + wave * 32;
  const int nBase = blockIdx.x * 64;

  v8f acc[2][4];
#pragma unroll
  for (int u = 0; u < 2; ++u)
#pragma unroll
    for (int t = 0; t < 4; ++t) acc[u][t] = vzero8f();

  const half_t* aRow0 = A + (size_t)(mBase + mLoc) * K;
  const half_t* aRow1 = A + (size_t)(mBase + 16 + mLoc) * K;
  const half_t* wRow  = W + (size_t)(nBase + mLoc) * K;

  for (int k0 = 0; k0 < K; k0 += 32) {
    if (k0 + 128 < K) {
      __builtin_prefetch((const void*)(aRow0 + k0 + 128), 0, 0);  // global_prefetch_b8
      __builtin_prefetch((const void*)(wRow + k0 + 128), 0, 0);
    }
    const v16h a0 = make_frag(aRow0 + k0 + hi * 8, aRow0 + k0 + 16 + hi * 8);
    const v16h a1 = make_frag(aRow1 + k0 + hi * 8, aRow1 + k0 + 16 + hi * 8);
#pragma unroll
    for (int t = 0; t < 4; ++t) {
      const half_t* bp = wRow + (size_t)(t * 16) * K + k0 + hi * 8;
      const v16h bfr = make_frag(bp, bp + 16);
      acc[0][t] = __builtin_amdgcn_wmma_f32_16x16x32_f16(
          false, a0, false, bfr, (short)0, acc[0][t], false, false);
      acc[1][t] = __builtin_amdgcn_wmma_f32_16x16x32_f16(
          false, a1, false, bfr, (short)0, acc[1][t], false, false);
    }
  }

#pragma unroll
  for (int u = 0; u < 2; ++u) {
#pragma unroll
    for (int t = 0; t < 4; ++t) {
      const int n = nBase + t * 16 + mLoc;
      const float bv = bias[n];
#pragma unroll
      for (int r = 0; r < 8; ++r) {
        const int m = mBase + u * 16 + r + 8 * hi;
        float v = acc[u][t][r] + bv;
        const size_t idx = (size_t)m * N + n;
        if (EPI == 0) {
          outH[idx] = (half_t)v;
        } else if (EPI == 1) {
          v = 0.5f * v * (1.0f + erff(v * 0.70710678118654752f));  // exact GELU
          outH[idx] = (half_t)v;
        } else {
          outF[idx] = v + res[idx];
        }
      }
    }
  }
}

// ---------------------------------------------------------------------------
// V^T staging: vp (B,256,C) f16  ->  vT (B*NH, 64, 256) f16 (d-major per head)
// so PV B-fragments become two 16B runs per lane.
// ---------------------------------------------------------------------------
__global__ void transpose_v_kernel(const half_t* __restrict__ vp,
                                   half_t* __restrict__ vT) {
  size_t i = (size_t)blockIdx.x * 256 + threadIdx.x;
  const size_t total = (size_t)Bsz * NHEAD * HD * NTOK2;
  if (i >= total) return;
  const int m = (int)(i & 255);
  size_t r = i >> 8;
  const int d = (int)(r & 63); r >>= 6;
  const int h = (int)(r % NHEAD);
  const int b = (int)(r / NHEAD);
  vT[i] = vp[((size_t)(b * NTOK2 + m)) * Cdim + h * HD + d];
}

// ---------------------------------------------------------------------------
// Flash attention: block = 4 waves, each wave owns 16 query rows.
// grid = (16 q-blocks, 12 heads, 32 batch). Keys processed in 2 slabs of 128
// (K slab + V^T slab staged in LDS), 32-key WMMA chunks, online softmax in
// the D-fragment lane layout, epilogue writes x1 = x + attn_out (f32).
// ---------------------------------------------------------------------------
__global__ __launch_bounds__(128)
void attn_kernel(const half_t* __restrict__ qp, const half_t* __restrict__ kp,
                 const half_t* __restrict__ vT, const float* __restrict__ x,
                 float* __restrict__ x1) {
  __shared__ alignas(16) half_t sK[128 * 64];      // keys (token-major) 16 KB
  __shared__ alignas(16) half_t sVt[64 * 128];     // V^T (d-major)      16 KB
  __shared__ alignas(16) half_t sP[4][16 * 32];    // per-wave P stage    4 KB

  const int b = blockIdx.z, h = blockIdx.y, qb = blockIdx.x;
  const int tid = threadIdx.x, wave = tid >> 5, lane = tid & 31;
  const int mLoc = lane & 15, hi = lane >> 4;
  const int qRow = qb * 64 + wave * 16;

  // Q fragments (16x64, two 32-wide k-steps), reused across all keys.
  const half_t* qptr = qp + ((size_t)(b * NTOK + qRow + mLoc)) * Cdim + h * HD;
  const v16h aq0 = make_frag(qptr + hi * 8,      qptr + 16 + hi * 8);
  const v16h aq1 = make_frag(qptr + 32 + hi * 8, qptr + 48 + hi * 8);

  v8f accO[4];
#pragma unroll
  for (int t = 0; t < 4; ++t) accO[t] = vzero8f();
  float rowM[8], rowL[8];
#pragma unroll
  for (int r = 0; r < 8; ++r) { rowM[r] = -1e30f; rowL[r] = 0.0f; }

  for (int slab = 0; slab < 2; ++slab) {
    __syncthreads();  // protect LDS reuse across slabs
    {   // stage K slab: 128 tokens x 64 d
      const int rr = tid;  // 128 threads, one row each
      const half_t* src = kp + ((size_t)(b * NTOK2 + slab * 128 + rr)) * Cdim + h * HD;
#pragma unroll
      for (int j = 0; j < 8; ++j)
        *(v8h*)&sK[rr * 64 + j * 8] = *(const v8h*)(src + j * 8);
    }
    {   // stage V^T slab: 64 d x 128 tokens
      const int d = tid >> 1, half128 = tid & 1;
      const half_t* src = vT + ((size_t)(b * NHEAD + h)) * HD * NTOK2 +
                          (size_t)d * NTOK2 + slab * 128 + half128 * 64;
#pragma unroll
      for (int j = 0; j < 8; ++j)
        *(v8h*)&sVt[d * 128 + half128 * 64 + j * 8] = *(const v8h*)(src + j * 8);
    }
    __syncthreads();

    for (int ck = 0; ck < 4; ++ck) {           // 32 keys per chunk
      v8f s0 = vzero8f(), s1 = vzero8f();
#pragma unroll
      for (int ks = 0; ks < 2; ++ks) {         // K dim (d) in 2 steps of 32
        const v16h a = ks ? aq1 : aq0;
        const half_t* bp0 = &sK[(ck * 32 + mLoc) * 64 + ks * 32 + hi * 8];
        const half_t* bp1 = &sK[(ck * 32 + 16 + mLoc) * 64 + ks * 32 + hi * 8];
        s0 = __builtin_amdgcn_wmma_f32_16x16x32_f16(
            false, a, false, make_frag(bp0, bp0 + 16), (short)0, s0, false, false);
        s1 = __builtin_amdgcn_wmma_f32_16x16x32_f16(
            false, a, false, make_frag(bp1, bp1 + 16), (short)0, s1, false, false);
      }
      // Online softmax (rows live across 16-lane halves in D layout).
      float pr0[8], pr1[8], corr[8];
#pragma unroll
      for (int r = 0; r < 8; ++r) {
        const float v0 = s0[r] * 0.125f, v1 = s1[r] * 0.125f;   // 1/sqrt(64)
        float mv = fmaxf(v0, v1);
#pragma unroll
        for (int msk = 1; msk < 16; msk <<= 1) mv = fmaxf(mv, __shfl_xor(mv, msk, 32));
        const float mNew = fmaxf(rowM[r], mv);
        corr[r] = expf(rowM[r] - mNew);
        const float p0 = expf(v0 - mNew), p1 = expf(v1 - mNew);
        float rs = p0 + p1;
#pragma unroll
        for (int msk = 1; msk < 16; msk <<= 1) rs += __shfl_xor(rs, msk, 32);
        rowL[r] = rowL[r] * corr[r] + rs;
        rowM[r] = mNew;
        pr0[r] = p0; pr1[r] = p1;
      }
#pragma unroll
      for (int t = 0; t < 4; ++t)
#pragma unroll
        for (int r = 0; r < 8; ++r) accO[t][r] *= corr[r];

      // Transpose P (D layout -> A layout) through wave-private LDS.
#pragma unroll
      for (int r = 0; r < 8; ++r) {
        const int m = r + 8 * hi;
        sP[wave][m * 32 + mLoc]      = (half_t)pr0[r];
        sP[wave][m * 32 + 16 + mLoc] = (half_t)pr1[r];
      }
      const half_t* pp = &sP[wave][mLoc * 32];
      const v16h ap = make_frag(pp + hi * 8, pp + 16 + hi * 8);

      // O += P(16x32) @ V(32x64)
#pragma unroll
      for (int t = 0; t < 4; ++t) {
        const half_t* bp = &sVt[(t * 16 + mLoc) * 128 + ck * 32 + hi * 8];
        accO[t] = __builtin_amdgcn_wmma_f32_16x16x32_f16(
            false, ap, false, make_frag(bp, bp + 16), (short)0, accO[t], false, false);
      }
    }
  }

  // Epilogue: x1 = x + O / l   (heads partition the C columns)
#pragma unroll
  for (int t = 0; t < 4; ++t) {
#pragma unroll
    for (int r = 0; r < 8; ++r) {
      const int m = b * NTOK + qRow + r + 8 * hi;
      const int n = h * HD + t * 16 + mLoc;
      const size_t idx = (size_t)m * Cdim + n;
      x1[idx] = x[idx] + accO[t][r] / rowL[r];
    }
  }
}

// ---------------------------------------------------------------------------
// Host launcher
// ---------------------------------------------------------------------------
extern "C" void kernel_launch(void* const* d_in, const int* in_sizes, int n_in,
                              void* d_out, int out_size, void* d_ws, size_t ws_size,
                              hipStream_t stream) {
  (void)in_sizes; (void)n_in; (void)out_size;

  const float* x     = (const float*)d_in[0];
  const float* ln1_g = (const float*)d_in[1];
  const float* ln1_b = (const float*)d_in[2];
  // per-branch blocks: dw_w, dw_b, bn_g, bn_b, bn_m, bn_v, W, b
  const float* dw_w[3] = {(const float*)d_in[3],  (const float*)d_in[11], (const float*)d_in[19]};
  const float* dw_b[3] = {(const float*)d_in[4],  (const float*)d_in[12], (const float*)d_in[20]};
  const float* bn_g[3] = {(const float*)d_in[5],  (const float*)d_in[13], (const float*)d_in[21]};
  const float* bn_b[3] = {(const float*)d_in[6],  (const float*)d_in[14], (const float*)d_in[22]};
  const float* bn_m[3] = {(const float*)d_in[7],  (const float*)d_in[15], (const float*)d_in[23]};
  const float* bn_v[3] = {(const float*)d_in[8],  (const float*)d_in[16], (const float*)d_in[24]};
  const float* Wp[3]   = {(const float*)d_in[9],  (const float*)d_in[17], (const float*)d_in[25]};
  const float* bp[3]   = {(const float*)d_in[10], (const float*)d_in[18], (const float*)d_in[26]};
  const float* ln2_g = (const float*)d_in[27];
  const float* ln2_b = (const float*)d_in[28];
  const float* W1    = (const float*)d_in[29];
  const float* b1    = (const float*)d_in[30];
  const float* W2    = (const float*)d_in[31];
  const float* b2    = (const float*)d_in[32];

  // Workspace arenas (aliased by lifetime). Peak ~365 MB.
  const size_t SZ_BNC_H  = (size_t)Bsz * NTOK  * Cdim * 2;   // 50,331,648
  const size_t SZ_BN2C_H = (size_t)Bsz * NTOK2 * Cdim * 2;   // 12,582,912
  const size_t SZ_H1_H   = (size_t)Bsz * NTOK  * FFC  * 2;   // 201,326,592
  const size_t SZ_X1_F   = (size_t)Bsz * NTOK  * Cdim * 4;   // 100,663,296
  const size_t SZ_W_H    = (size_t)Cdim * Cdim * 2;
  const size_t SZ_W1_H   = (size_t)FFC  * Cdim * 2;

  const size_t o_xn  = 0;
  const size_t o_brq = o_xn  + SZ_BNC_H;
  const size_t o_brk = o_brq + SZ_BNC_H;
  const size_t o_brv = o_brk + SZ_BN2C_H;
  const size_t o_qp  = o_brv + SZ_BN2C_H;
  const size_t o_kp  = o_qp  + SZ_BNC_H;
  const size_t o_vp  = o_kp  + SZ_BN2C_H;
  const size_t o_vT  = o_vp  + SZ_BN2C_H;
  // Phase-2 aliases (all phase-1 tenants dead after attention):
  const size_t o_h16 = 0;                    // aliases xn
  const size_t o_h1  = SZ_BNC_H;             // aliases brq..vT (+tail)
  const size_t arenaA = (o_h1 + SZ_H1_H > o_vT + SZ_BN2C_H) ? (o_h1 + SZ_H1_H)
                                                            : (o_vT + SZ_BN2C_H);
  const size_t o_x1  = arenaA;
  const size_t o_w0  = o_x1 + SZ_X1_F;       // wq16, wk16, wv16, w116, w216
  const size_t need  = o_w0 + 3 * SZ_W_H + 2 * SZ_W1_H;
  if (ws_size < need) return;

  char* ws = (char*)d_ws;
  half_t* xn  = (half_t*)(ws + o_xn);
  half_t* brq = (half_t*)(ws + o_brq);
  half_t* brk = (half_t*)(ws + o_brk);
  half_t* brv = (half_t*)(ws + o_brv);
  half_t* qpb = (half_t*)(ws + o_qp);
  half_t* kpb = (half_t*)(ws + o_kp);
  half_t* vpb = (half_t*)(ws + o_vp);
  half_t* vTb = (half_t*)(ws + o_vT);
  half_t* h16 = (half_t*)(ws + o_h16);
  half_t* h1  = (half_t*)(ws + o_h1);
  float*  x1  = (float*)(ws + o_x1);
  half_t* wq16 = (half_t*)(ws + o_w0);
  half_t* wk16 = (half_t*)(ws + o_w0 + SZ_W_H);
  half_t* wv16 = (half_t*)(ws + o_w0 + 2 * SZ_W_H);
  half_t* w116 = (half_t*)(ws + o_w0 + 3 * SZ_W_H);
  half_t* w216 = (half_t*)(ws + o_w0 + 3 * SZ_W_H + SZ_W1_H);

  // 1) Stage weights to f16 (kept (N,K) row-major -> contiguous B-fragments).
  const int nW  = Cdim * Cdim;           // 589,824
  const int nW1 = FFC * Cdim;            // 2,359,296
  f32_to_f16_kernel<<<(nW  + 255) / 256, 256, 0, stream>>>(Wp[0], wq16, nW);
  f32_to_f16_kernel<<<(nW  + 255) / 256, 256, 0, stream>>>(Wp[1], wk16, nW);
  f32_to_f16_kernel<<<(nW  + 255) / 256, 256, 0, stream>>>(Wp[2], wv16, nW);
  f32_to_f16_kernel<<<(nW1 + 255) / 256, 256, 0, stream>>>(W1, w116, nW1);
  f32_to_f16_kernel<<<(nW1 + 255) / 256, 256, 0, stream>>>(W2, w216, nW1);

  // 2) LN1 -> xn (f16)
  ln_kernel<<<Bsz * NTOK, 256, 0, stream>>>(x, ln1_g, ln1_b, xn);

  // 3) Depthwise branches
  dw_kernel<1><<<dim3(3, 1024, Bsz), 256, 0, stream>>>(
      xn, dw_w[0], dw_b[0], bn_g[0], bn_b[0], bn_m[0], bn_v[0], brq);
  dw_kernel<2><<<dim3(3, 256, Bsz), 256, 0, stream>>>(
      xn, dw_w[1], dw_b[1], bn_g[1], bn_b[1], bn_m[1], bn_v[1], brk);
  dw_kernel<2><<<dim3(3, 256, Bsz), 256, 0, stream>>>(
      xn, dw_w[2], dw_b[2], bn_g[2], bn_b[2], bn_m[2], bn_v[2], brv);

  // 4) Q/K/V projections (WMMA GEMMs), f16 outputs. Block tile = 256 rows.
  gemm_wmma_kernel<0><<<dim3(Cdim / 64, (Bsz * NTOK) / 256), 256, 0, stream>>>(
      brq, wq16, bp[0], nullptr, qpb, nullptr, Bsz * NTOK, Cdim, Cdim);
  gemm_wmma_kernel<0><<<dim3(Cdim / 64, (Bsz * NTOK2) / 256), 256, 0, stream>>>(
      brk, wk16, bp[1], nullptr, kpb, nullptr, Bsz * NTOK2, Cdim, Cdim);
  gemm_wmma_kernel<0><<<dim3(Cdim / 64, (Bsz * NTOK2) / 256), 256, 0, stream>>>(
      brv, wv16, bp[2], nullptr, vpb, nullptr, Bsz * NTOK2, Cdim, Cdim);

  // 5) V^T per head
  {
    const size_t total = (size_t)Bsz * NHEAD * HD * NTOK2;
    transpose_v_kernel<<<(unsigned)((total + 255) / 256), 256, 0, stream>>>(vpb, vTb);
  }

  // 6) Flash attention + residual -> x1 (f32)
  attn_kernel<<<dim3(NTOK / 64, NHEAD, Bsz), 128, 0, stream>>>(qpb, kpb, vTb, x, x1);

  // 7) LN2 -> h16 (f16)
  ln_kernel<<<Bsz * NTOK, 256, 0, stream>>>(x1, ln2_g, ln2_b, h16);

  // 8) MLP: GELU GEMM then projection + residual -> d_out (f32)
  gemm_wmma_kernel<1><<<dim3(FFC / 64, (Bsz * NTOK) / 256), 256, 0, stream>>>(
      h16, w116, b1, nullptr, h1, nullptr, Bsz * NTOK, FFC, Cdim);
  gemm_wmma_kernel<2><<<dim3(Cdim / 64, (Bsz * NTOK) / 256), 256, 0, stream>>>(
      h1, w216, b2, x1, nullptr, (float*)d_out, Bsz * NTOK, Cdim, FFC);
}
